// WaveletField_74552042324251
// MI455X (gfx1250) — matrix operand order repaired and checked by reading, not compiled, not run on hardware
//
#include <hip/hip_runtime.h>
#include <hip/hip_bf16.h>

typedef __attribute__((ext_vector_type(16))) _Float16 v16h;
typedef __attribute__((ext_vector_type(8)))  _Float16 v8h;
typedef __attribute__((ext_vector_type(8)))  float    v8f;

// ---------------- workspace layout ----------------
// floats: hi planes | lo planes | enc[8192*16]
// then f16 weights at byte offset WT_BYTE
// plane float sizes: {524288,524288,409600,524288,409600,409600}
__device__ __host__ constexpr size_t HI_OFF[6] = {0ul, 524288ul, 1048576ul, 1458176ul, 1982464ul, 2392064ul};
__device__ __host__ constexpr size_t LO_OFF[6] = {2801664ul, 3325952ul, 3850240ul, 4259840ul, 4784128ul, 5193728ul};
#define ENC_F_OFF  5603328ul               // floats
#define WT_BYTE    22937600ul              // (5603328 + 8192*16) * 4
// f16 weight region offsets (halfs)
#define WS1_OFF 0      // [64][64]
#define WS2_OFF 4096   // [16][64]
#define WC1_OFF 5120   // [64][32]
#define WC2_OFF 7168   // [64][64]
#define WC3_OFF 11264  // [16][64]

// ---------------- IDWT: yl[C,h,w], yh[C,3,h,w] -> hi/lo [2h][2w][32] ----------------
__global__ void idwt_kernel(const float* __restrict__ yl, const float* __restrict__ yh,
                            float* __restrict__ hi, float* __restrict__ lo, int h, int w) {
    int tid = blockIdx.x * blockDim.x + threadIdx.x;
    int total = 32 * h * w;
    if (tid >= total) return;
    int c = tid & 31;
    int rest = tid >> 5;
    int x = rest % w, y = rest / w;
    float a  = yl[(c * h + y) * w + x];
    float lh = yh[((c * 3 + 0) * h + y) * w + x];
    float hl = yh[((c * 3 + 1) * h + y) * w + x];
    float hh = yh[((c * 3 + 2) * h + y) * w + x];
    float ee = 0.5f * (a + lh + hl + hh);
    float eo = 0.5f * (a + lh - hl - hh);
    float oe = 0.5f * (a - lh + hl - hh);
    float oo = 0.5f * (a - lh - hl + hh);
    int W = 2 * w;
    size_t b00 = ((size_t)(2 * y) * W + 2 * x) * 32 + c;
    size_t rowp = (size_t)W * 32;
    hi[b00] = ee; hi[b00 + 32] = eo; hi[b00 + rowp] = oe; hi[b00 + rowp + 32] = oo;
    float ha = 0.5f * a;
    lo[b00] = ha; lo[b00 + 32] = ha; lo[b00 + rowp] = ha; lo[b00 + rowp + 32] = ha;
}

// ---------------- SH16 per-ray encoding ----------------
__global__ void sh_kernel(const float* __restrict__ dirs, float* __restrict__ enc, int R) {
    int r = blockIdx.x * blockDim.x + threadIdx.x;
    if (r >= R) return;
    float x = dirs[r * 3 + 0], y = dirs[r * 3 + 1], z = dirs[r * 3 + 2];
    float inv = rsqrtf(x * x + y * y + z * z);
    x *= inv; y *= inv; z *= inv;
    float xx = x * x, yy = y * y, zz = z * z;
    float* e = enc + (size_t)r * 16;
    e[0]  = 0.28209479177387814f;
    e[1]  = -0.4886025119029199f * y;
    e[2]  =  0.4886025119029199f * z;
    e[3]  = -0.4886025119029199f * x;
    e[4]  =  1.0925484305920792f * x * y;
    e[5]  = -1.0925484305920792f * y * z;
    e[6]  =  0.31539156525252005f * (3.0f * zz - 1.0f);
    e[7]  = -1.0925484305920792f * x * z;
    e[8]  =  0.5462742152960396f * (xx - yy);
    e[9]  = -0.5900435899266435f * y * (3.0f * xx - yy);
    e[10] =  2.890611442640554f * x * y * z;
    e[11] = -0.4570457994644658f * y * (5.0f * zz - 1.0f);
    e[12] =  0.3731763325901154f * z * (5.0f * zz - 3.0f);
    e[13] = -0.4570457994644658f * x * (5.0f * zz - 1.0f);
    e[14] =  1.445305721320277f * z * (xx - yy);
    e[15] = -0.5900435899266435f * x * (xx - 3.0f * yy);
}

// ---------------- weight transpose + f32->f16 ----------------
__global__ void wconv_kernel(const float* __restrict__ Wsig1, const float* __restrict__ Wsig2,
                             const float* __restrict__ Wc1,   const float* __restrict__ Wc2,
                             const float* __restrict__ Wc3,   _Float16* __restrict__ wt) {
    int idx = blockIdx.x * blockDim.x + threadIdx.x;
    if (idx >= 12288) return;
    float v = 0.0f;
    if (idx < 4096) {                       // Wsig1T [64][64]
        int n = idx >> 6, k = idx & 63;
        v = Wsig1[k * 64 + n];
    } else if (idx < 5120) {                // Wsig2T [16][64]
        int i = idx - 4096; int n = i >> 6, k = i & 63;
        v = Wsig2[k * 16 + n];
    } else if (idx < 7168) {                // Wc1T [64][32], K padded 31->32
        int i = idx - 5120; int n = i >> 5, k = i & 31;
        v = (k < 31) ? Wc1[k * 64 + n] : 0.0f;
    } else if (idx < 11264) {               // Wc2T [64][64]
        int i = idx - 7168; int n = i >> 6, k = i & 63;
        v = Wc2[k * 64 + n];
    } else {                                // Wc3T [16][64], N padded 3->16
        int i = idx - 11264; int n = i >> 6, k = i & 63;
        v = (n < 3) ? Wc3[k * 3 + n] : 0.0f;
    }
    wt[idx] = (_Float16)v;
}

// ---------------- WMMA fragment helpers ----------------
__device__ __forceinline__ v8f wmma32(v16h a, v16h b, v8f c) {
    return __builtin_amdgcn_wmma_f32_16x16x32_f16(false, a, false, b, (short)0, c, false, false);
}

// A fragment (16x32 f16) from LDS row-major [16][stride], K base kb (0 or 32)
__device__ __forceinline__ v16h load_a_frag(const _Float16* lds, int stride, int kb) {
    int lane = threadIdx.x & 31;
    int m = lane & 15;
    int k0 = kb + ((lane >> 4) << 3);           // +0 or +8
    const _Float16* p = lds + m * stride + k0;
    union { v16h v; v8h h[2]; } u;
    u.h[0] = *(const v8h*)(p);                  // K = k0 .. k0+7   -> VGPR 0-3
    u.h[1] = *(const v8h*)(p + 16);             // K = k0+16..k0+23 -> VGPR 4-7
    return u.v;
}

// B fragment (32x16 f16) from transposed f16 weights Wt[N][Kpad]
__device__ __forceinline__ v16h load_b_frag(const _Float16* wt, int Kpad, int nt, int kb) {
    int lane = threadIdx.x & 31;
    int col = nt * 16 + (lane & 15);
    int k0 = kb + ((lane >> 4) << 3);           // +0 or +8
    const _Float16* p = wt + col * Kpad + k0;
    union { v16h v; v8h h[2]; } u;
    u.h[0] = *(const v8h*)(p);
    u.h[1] = *(const v8h*)(p + 16);
    return u.v;
}

// C (VGPR r holds element M = r + 8*(lane>=16), N = lane&15) -> relu -> LDS f16
__device__ __forceinline__ void store_c_relu(v8f c, _Float16* lds, int stride, int nb) {
    int lane = threadIdx.x & 31;
    int n = nb + (lane & 15);
    int mb = (lane >> 4) << 3;
#pragma unroll
    for (int r = 0; r < 8; ++r)
        lds[(mb + r) * stride + n] = (_Float16)fmaxf(c[r], 0.0f);
}

// ---------------- main fused kernel ----------------
__global__ __launch_bounds__(256)
void wavefield_main(const float* __restrict__ pts, const float* __restrict__ timestamps,
                    const float* __restrict__ planes, const float* __restrict__ enc_g,
                    const _Float16* __restrict__ wt, float* __restrict__ out) {
    constexpr int PJ[6] = {0, 0, 0, 1, 1, 2};
    constexpr int PK[6] = {1, 2, 3, 2, 3, 3};
    constexpr int PW[6] = {128, 128, 100, 128, 100, 100};

    __shared__ __align__(16) _Float16 s_a [8][16 * 72];   // feats / layer-3 out
    __shared__ __align__(16) _Float16 s_b [8][16 * 72];   // layer-1 out / layer-4 out
    __shared__ __align__(16) _Float16 s_ci[8][16 * 40];   // color-net input (K=32, pad 40)

    const int lane = threadIdx.x & 31;
    const int wv   = threadIdx.x >> 5;
    const int tile = blockIdx.x * 8 + wv;
    const int base = tile * 16;                 // 16 points per tile, same ray
    const int ray  = base >> 6;

    _Float16* feats = s_a[wv];
    _Float16* hbuf  = s_b[wv];
    _Float16* ci    = s_ci[wv];

    // ---------- phase 1: wavelet feature sampling (lane = channel) ----------
    const float t = timestamps[ray] * 2.0f - 1.0f;
#pragma unroll 1
    for (int p = 0; p < 16; ++p) {
        const int q = base + p;
        float p4[4];
        p4[0] = pts[(size_t)q * 3 + 0];
        p4[1] = pts[(size_t)q * 3 + 1];
        p4[2] = pts[(size_t)q * 3 + 2];
        p4[3] = t;
        float fh = 1.0f, fl = 1.0f;
#pragma unroll
        for (int pl = 0; pl < 6; ++pl) {
            const float vj = p4[PJ[pl]], vk = p4[PK[pl]];
            const int W = PW[pl];
            float yy = (vj + 1.0f) * 0.5f * 127.0f;
            float xx = (vk + 1.0f) * 0.5f * (float)(W - 1);
            float y0f = fminf(fmaxf(floorf(yy), 0.0f), 127.0f);
            float x0f = fminf(fmaxf(floorf(xx), 0.0f), (float)(W - 1));
            int y0 = (int)y0f, x0 = (int)x0f;
            int y1 = (y0 + 1 < 127) ? y0 + 1 : 127;
            int x1 = (x0 + 1 < W - 1) ? x0 + 1 : W - 1;
            float wy = fminf(fmaxf(yy - y0f, 0.0f), 1.0f);
            float wx = fminf(fmaxf(xx - x0f, 0.0f), 1.0f);
            float w00 = (1.0f - wy) * (1.0f - wx), w01 = (1.0f - wy) * wx;
            float w10 = wy * (1.0f - wx),          w11 = wy * wx;
            const float* hi = planes + HI_OFF[pl];
            const float* lo = planes + LO_OFF[pl];
            int i00 = ((y0 * W + x0) << 5) + lane;
            int i01 = ((y0 * W + x1) << 5) + lane;
            int i10 = ((y1 * W + x0) << 5) + lane;
            int i11 = ((y1 * W + x1) << 5) + lane;
            fh *= hi[i00] * w00 + hi[i01] * w01 + hi[i10] * w10 + hi[i11] * w11;
            fl *= lo[i00] * w00 + lo[i01] * w01 + lo[i10] * w10 + lo[i11] * w11;
        }
        feats[p * 72 + lane]      = (_Float16)fh;   // cols 0..31  = feat_hi
        feats[p * 72 + 32 + lane] = (_Float16)fl;   // cols 32..63 = feat_lo
    }
    __syncthreads();

    // ---------- layer 1: feats[16,64] @ Wsig1[64,64] -> relu -> hbuf ----------
    {
        v16h a0 = load_a_frag(feats, 72, 0);
        v16h a1 = load_a_frag(feats, 72, 32);
        const _Float16* w1 = wt + WS1_OFF;
#pragma unroll
        for (int nt = 0; nt < 4; ++nt) {
            v8f c = {};
            c = wmma32(a0, load_b_frag(w1, 64, nt, 0),  c);
            c = wmma32(a1, load_b_frag(w1, 64, nt, 32), c);
            store_c_relu(c, hbuf, 72, nt * 16);
        }
    }
    __syncthreads();

    // ---------- layer 2: h[16,64] @ Wsig2[64,16]; geo->ci, density->out ----------
    {
        v16h a0 = load_a_frag(hbuf, 72, 0);
        v16h a1 = load_a_frag(hbuf, 72, 32);
        const _Float16* w2 = wt + WS2_OFF;
        v8f c = {};
        c = wmma32(a0, load_b_frag(w2, 64, 0, 0),  c);
        c = wmma32(a1, load_b_frag(w2, 64, 0, 32), c);
        const int n  = lane & 15;
        const int mb = (lane >> 4) << 3;
        if (n < 15) {
#pragma unroll
            for (int r = 0; r < 8; ++r)
                ci[(mb + r) * 40 + 16 + n] = (_Float16)c[r];     // geo cols 16..30
        } else {
#pragma unroll
            for (int r = 0; r < 8; ++r) {
                float d = __expf(fminf(fmaxf(c[r], -15.0f), 15.0f));
                out[(size_t)(base + mb + r) * 4 + 3] = d;        // density
                ci[(mb + r) * 40 + 31] = (_Float16)0.0f;         // K pad col
            }
        }
        // SH encoding (tile-uniform per ray): cols 0..15
        float ev = enc_g[(size_t)ray * 16 + n];
#pragma unroll
        for (int r = 0; r < 8; ++r)
            ci[(mb + r) * 40 + n] = (_Float16)ev;
    }
    __syncthreads();

    // ---------- layer 3: ci[16,32] @ Wc1[32,64] -> relu -> s_a ----------
    {
        v16h a0 = load_a_frag(ci, 40, 0);
        const _Float16* w3 = wt + WC1_OFF;
#pragma unroll
        for (int nt = 0; nt < 4; ++nt) {
            v8f c = {};
            c = wmma32(a0, load_b_frag(w3, 32, nt, 0), c);
            store_c_relu(c, feats, 72, nt * 16);
        }
    }
    __syncthreads();

    // ---------- layer 4: h[16,64] @ Wc2[64,64] -> relu -> s_b ----------
    {
        v16h a0 = load_a_frag(feats, 72, 0);
        v16h a1 = load_a_frag(feats, 72, 32);
        const _Float16* w4 = wt + WC2_OFF;
#pragma unroll
        for (int nt = 0; nt < 4; ++nt) {
            v8f c = {};
            c = wmma32(a0, load_b_frag(w4, 64, nt, 0),  c);
            c = wmma32(a1, load_b_frag(w4, 64, nt, 32), c);
            store_c_relu(c, hbuf, 72, nt * 16);
        }
    }
    __syncthreads();

    // ---------- layer 5: h[16,64] @ Wc3[64,16pad] -> sigmoid -> rgb ----------
    {
        v16h a0 = load_a_frag(hbuf, 72, 0);
        v16h a1 = load_a_frag(hbuf, 72, 32);
        const _Float16* w5 = wt + WC3_OFF;
        v8f c = {};
        c = wmma32(a0, load_b_frag(w5, 64, 0, 0),  c);
        c = wmma32(a1, load_b_frag(w5, 64, 0, 32), c);
        const int n  = lane & 15;
        const int mb = (lane >> 4) << 3;
        if (n < 3) {
#pragma unroll
            for (int r = 0; r < 8; ++r) {
                float s = 1.0f / (1.0f + __expf(-c[r]));
                out[(size_t)(base + mb + r) * 4 + n] = s;
            }
        }
    }
}

// ---------------- host launch ----------------
extern "C" void kernel_launch(void* const* d_in, const int* in_sizes, int n_in,
                              void* d_out, int out_size, void* d_ws, size_t ws_size,
                              hipStream_t stream) {
    const float* pts   = (const float*)d_in[0];
    const float* dirs  = (const float*)d_in[1];
    const float* ts    = (const float*)d_in[2];
    const float* Wsig1 = (const float*)d_in[15];
    const float* Wsig2 = (const float*)d_in[16];
    const float* Wc1   = (const float*)d_in[17];
    const float* Wc2   = (const float*)d_in[18];
    const float* Wc3   = (const float*)d_in[19];

    float*     wsf = (float*)d_ws;
    _Float16*  wt  = (_Float16*)((char*)d_ws + WT_BYTE);

    const int R = in_sizes[2];           // 8192 rays
    const int N = in_sizes[0] / 3;       // 524288 points

    static const int hh[6] = {64, 64, 64, 64, 64, 64};
    static const int ww[6] = {64, 64, 50, 64, 50, 50};
    for (int i = 0; i < 6; ++i) {
        int total = 32 * hh[i] * ww[i];
        idwt_kernel<<<(total + 255) / 256, 256, 0, stream>>>(
            (const float*)d_in[3 + 2 * i], (const float*)d_in[4 + 2 * i],
            wsf + HI_OFF[i], wsf + LO_OFF[i], hh[i], ww[i]);
    }
    sh_kernel<<<(R + 255) / 256, 256, 0, stream>>>(dirs, wsf + ENC_F_OFF, R);
    wconv_kernel<<<48, 256, 0, stream>>>(Wsig1, Wsig2, Wc1, Wc2, Wc3, wt);

    const int tiles = N / 16;            // 32768 tiles, 8 waves/block
    wavefield_main<<<tiles / 8, 256, 0, stream>>>(pts, ts, wsf, wsf + ENC_F_OFF, wt, (float*)d_out);
}
